// NKATAttention_2121713844858
// MI455X (gfx1250) — matrix operand order repaired and checked by reading, not compiled, not run on hardware
//
#include <hip/hip_runtime.h>
#include <hip/hip_bf16.h>

// Problem constants (match reference)
#define B_ 2
#define N_ 1024
#define C_ 1024
#define H_ 16
#define HD_ 64
#define TOPK_ 64
#define TOPP_ 0.9f
#define QK_SCALE 0.15625f      // HD^-0.5 / TEMPERATURE = 0.125 / 0.8
#define LAYER_COEFF 0.45125f   // 0.5 * 0.95^2

typedef __attribute__((ext_vector_type(16))) __bf16 v16bf;
typedef __attribute__((ext_vector_type(8)))  __bf16 v8bf;
typedef __attribute__((ext_vector_type(8)))  float  v8f;
typedef __attribute__((ext_vector_type(4)))  unsigned int u32x4;
typedef __attribute__((ext_vector_type(8)))  int i32x8;
typedef __attribute__((ext_vector_type(4)))  int i32x4;

// ---- Tensor Data Mover availability (probe-verified builtin) ----
#if defined(__has_builtin)
#if __has_builtin(__builtin_amdgcn_tensor_load_to_lds) && \
    __has_builtin(__builtin_amdgcn_s_wait_tensorcnt)
#define HAVE_TDM 1
#endif
#endif
#ifndef HAVE_TDM
#define HAVE_TDM 0
#endif

__device__ __forceinline__ __bf16 f2bf(float f) {
  union { float f; unsigned u; } a; a.f = f;
  unsigned r = a.u + 0x7fffu + ((a.u >> 16) & 1u);   // round-to-nearest-even
  unsigned short h = (unsigned short)(r >> 16);
  union { unsigned short s; __bf16 b; } c; c.s = h;
  return c.b;
}

__device__ __forceinline__ v16bf cat16(v8bf lo, v8bf hi) {
  return __builtin_shufflevector(lo, hi, 0, 1, 2, 3, 4, 5, 6, 7,
                                 8, 9, 10, 11, 12, 13, 14, 15);
}

// ---- WMMA fragment loaders (CDNA5 wave32 layouts, 16x16x32 bf16) ----
// A (16xK tile, row-major, 16-bit): lane<16 row=lane holds K {0..7,16..23};
// lane>=16 same row, K {8..15,24..31}. Two contiguous 16B chunks per lane.
__device__ __forceinline__ v16bf load_a_bf16(const __bf16* A, int lda, int lane) {
  const __bf16* r = A + (lane & 15) * lda + ((lane >> 4) << 3);
  v8bf lo = *(const v8bf*)r;
  v8bf hi = *(const v8bf*)(r + 16);
  return cat16(lo, hi);
}

// B (32x16 tile) from a K-contiguous ("transposed") layout: element (k,col) at
// base[col*ldk + k]. lane<16: col=lane, K {0..7,16..23}; lane>=16: K {8..15,24..31}.
// Two contiguous 16B chunks per lane -> two b128 loads (global or ds).
__device__ __forceinline__ v16bf load_bT(const __bf16* base, int ldk, int lane) {
  const __bf16* r = base + (lane & 15) * ldk + ((lane >> 4) << 3);
  v8bf lo = *(const v8bf*)r;
  v8bf hi = *(const v8bf*)(r + 16);
  return cat16(lo, hi);
}

__device__ __forceinline__ v8f wmma_bf16(v16bf a, v16bf b, v8f c) {
  return __builtin_amdgcn_wmma_f32_16x16x32_bf16(false, a, false, b, (short)0, c,
                                                 false, false);
}

// ---------------- fp32 -> bf16 conversion ----------------
__global__ void __launch_bounds__(256) f32_to_bf16_kernel(const float* __restrict__ in,
                                                          __bf16* __restrict__ out, int n) {
  for (int i = blockIdx.x * blockDim.x + threadIdx.x; i < n; i += gridDim.x * blockDim.x)
    out[i] = f2bf(in[i]);
}

// ---------------- NKAT per-token row scale ----------------
__global__ void __launch_bounds__(256) theta_kernel(const float* __restrict__ x,
                                                    const float* __restrict__ alpha,
                                                    const float* __restrict__ beta,
                                                    float* __restrict__ rowScale) {
  __shared__ float rs[8], rs2[8];
  const int t = blockIdx.x;
  const float* xr = x + (size_t)t * C_;
  float s = 0.f, s2 = 0.f;
  for (int j = threadIdx.x; j < C_; j += 256) { float v = xr[j]; s += v; s2 += v * v; }
  for (int o = 16; o; o >>= 1) { s += __shfl_xor(s, o); s2 += __shfl_xor(s2, o); }
  const int lane = threadIdx.x & 31, wave = threadIdx.x >> 5;
  if (lane == 0) { rs[wave] = s; rs2[wave] = s2; }
  __syncthreads();
  if (threadIdx.x == 0) {
    float S = 0.f, S2 = 0.f;
    for (int w = 0; w < 8; ++w) { S += rs[w]; S2 += rs2[w]; }
    float mean = S * (1.f / C_);
    float var = S2 * (1.f / C_) - mean * mean;
    float th = alpha[0] * tanhf(mean) + beta[0] * (1.f / (1.f + __expf(-var)));
    rowScale[t] = 1.f + LAYER_COEFF * th;
  }
}

// Cooperative stage of a 32(K) x 128(N) weight tile into LDS, TRANSPOSED to
// sBT[col][k] (k fastest, ldk=32) so B fragments are contiguous per lane.
// Each thread: two coalesced b128 global loads (adjacent k rows), 8 packed b32
// ds stores.
__device__ __forceinline__ void stage_bT(const __bf16* __restrict__ W, int ldw,
                                         int k0, int nblk, __bf16* sBT, int tid) {
  const int k2 = (tid >> 4) << 1;          // even k row: 0,2,..,30
  const int colg = (tid & 15) << 3;        // 8 columns per thread
  const __bf16* s0 = W + (size_t)(k0 + k2) * ldw + nblk + colg;
  v8bf r0 = *(const v8bf*)s0;
  v8bf r1 = *(const v8bf*)(s0 + ldw);
#pragma unroll
  for (int i = 0; i < 8; ++i) {
    union { __bf16 h[2]; unsigned u; } p;
    p.h[0] = r0[i]; p.h[1] = r1[i];
    *(unsigned*)&sBT[(colg + i) * 32 + k2] = p.u;
  }
}

// ---------------- QKV projection GEMM (2048x3072, K=1024) ----------------
// 8 waves/WG, wave tile 32x64 (2x4 accs), WG tile 128x128. B staged transposed.
__global__ void __launch_bounds__(256) qkv_gemm_kernel(const __bf16* __restrict__ X,
                                                       const __bf16* __restrict__ W,
                                                       __bf16* __restrict__ Q,
                                                       __bf16* __restrict__ K,
                                                       __bf16* __restrict__ Vt) {
  __shared__ __align__(16) __bf16 sBT[128 * 32];   // 8KB, [col][k]
  const int lane = threadIdx.x & 31;
  const int wave = threadIdx.x >> 5;
  const int m0 = blockIdx.y * 128 + (wave >> 1) * 32;
  const int nblk = blockIdx.x * 128;
  const int nloc = (wave & 1) * 64;
  v8f acc[2][4] = {};
  for (int k0 = 0; k0 < C_; k0 += 32) {
    stage_bT(W, 3 * C_, k0, nblk, sBT, threadIdx.x);
    if (k0 + 32 < C_)
      __builtin_prefetch(W + (size_t)(k0 + 32 + ((threadIdx.x >> 4) << 1)) * (3 * C_)
                         + nblk + ((threadIdx.x & 15) << 3), 0, 1);
    __syncthreads();
    v16bf a0 = load_a_bf16(X + (size_t)m0 * C_ + k0, C_, lane);
    v16bf a1 = load_a_bf16(X + (size_t)(m0 + 16) * C_ + k0, C_, lane);
#pragma unroll
    for (int j = 0; j < 4; ++j) {
      v16bf bj = load_bT(sBT + (nloc + j * 16) * 32, 32, lane);
      acc[0][j] = wmma_bf16(a0, bj, acc[0][j]);
      acc[1][j] = wmma_bf16(a1, bj, acc[1][j]);
    }
    __syncthreads();
  }
  // Epilogue: scatter to head-major q/k and TRANSPOSED v^T[b][h][d][n];
  // fold score scaling into q.
#pragma unroll
  for (int i = 0; i < 2; ++i)
#pragma unroll
    for (int j = 0; j < 4; ++j)
#pragma unroll
      for (int r = 0; r < 8; ++r) {
        int gm = m0 + i * 16 + r + ((lane >> 4) << 3);
        int gc = nblk + nloc + j * 16 + (lane & 15);
        int b = gm >> 10, n = gm & (N_ - 1);
        int s = gc >> 10, rem = gc & (C_ - 1);
        int h = rem >> 6, d = rem & (HD_ - 1);
        float v = acc[i][j][r];
        if (s == 0) v *= QK_SCALE;
        size_t bhp = (size_t)(b * H_ + h);
        if (s == 2) {
          Vt[(bhp * HD_ + d) * N_ + n] = f2bf(v);
        } else {
          __bf16* dst = (s == 0) ? Q : K;
          dst[(bhp * N_ + n) * HD_ + d] = f2bf(v);
        }
      }
}

__device__ __forceinline__ unsigned f2ord(float f) {
  unsigned u = __float_as_uint(f);
  return (u & 0x80000000u) ? ~u : (u | 0x80000000u);   // monotone order-preserving
}

// ---------------- attention: scores -> topk/topp -> softmax -> P@V ----------------
// grid = (N/16, B*H); block = 256 (8 waves).
// Dynamic LDS: 64KB f32 scores + 32KB bf16 probs + 128KB bf16 V^T panel.
__global__ void __launch_bounds__(256) attn_kernel(const __bf16* __restrict__ Q,
                                                   const __bf16* __restrict__ K,
                                                   const __bf16* __restrict__ Vt,
                                                   const float* __restrict__ rowScale,
                                                   __bf16* __restrict__ AOUT) {
  extern __shared__ char smem[];
  float*  sS  = (float*)smem;                          // 16 x 1024 f32 scores
  __bf16* sP  = (__bf16*)(smem + 16 * N_ * 4);         // 16 x 1024 bf16 probs
  __bf16* sVT = (__bf16*)(smem + 16 * N_ * 4 + 16 * N_ * 2);  // 64 x 1024 bf16 V^T
  __shared__ float sKept[8][128];
  const int lane = threadIdx.x & 31;
  const int wave = threadIdx.x >> 5;
  const int bh = blockIdx.y;
  const int b = bh >> 4;
  const int h = bh & 15;
  const int q0 = blockIdx.x * 16;
  const __bf16* Qp = Q + ((size_t)bh * N_ + q0) * HD_;
  const __bf16* Kp = K + (size_t)bh * N_ * HD_;
  const __bf16* Vp = Vt + (size_t)bh * HD_ * N_;       // 64 rows x 1024 cols

#if HAVE_TDM
  // Tensor Data Mover: DMA the 64x1024 bf16 V^T panel into LDS, overlapped
  // with score WMMAs + the top-k/top-p phase. D# per ISA 8.3/8.4.
  if (wave == 0) {
    unsigned long long ga = (unsigned long long)Vp;
    unsigned ldsOff = (unsigned)(unsigned long long)sVT;
    u32x4 g0;
    g0[0] = 1u;                                        // count=1 (valid), user mode
    g0[1] = ldsOff;                                    // lds_addr
    g0[2] = (unsigned)(ga & 0xffffffffu);              // global_addr lo
    g0[3] = (unsigned)((ga >> 32) & 0x01ffffffu) | (2u << 30);  // addr hi | type=2
    i32x8 g1;
    g1[0] = (int)(1u << 16);                           // data_size=1 -> 2 bytes
    g1[1] = (int)((unsigned)(N_ & 0xffff) << 16);      // tensor_dim0 = 1024 (lo16)
    g1[2] = (int)((unsigned)(HD_ & 0xffff) << 16);     // dim0 hi=0 | tensor_dim1 = 64
    g1[3] = (int)((unsigned)N_ << 16);                 // dim1 hi=0 | tile_dim0 = 1024
    g1[4] = (int)HD_;                                  // tile_dim1 = 64 | tile_dim2 = 0
    g1[5] = (int)N_;                                   // tensor_dim0_stride = 1024
    g1[6] = 0;
    g1[7] = 0;
    i32x4 gz = {0, 0, 0, 0};
#if __clang_major__ >= 23
    i32x8 gz8 = {0, 0, 0, 0, 0, 0, 0, 0};
    __builtin_amdgcn_tensor_load_to_lds(g0, g1, gz, gz, gz8, 0);
#else
    __builtin_amdgcn_tensor_load_to_lds(g0, g1, gz, gz, 0);
#endif
  }
#else
  // Fallback: cooperative coalesced copy of the (already transposed) V panel.
  for (int i = threadIdx.x; i < (N_ * HD_) / 8; i += 256)
    ((v8bf*)sVT)[i] = ((const v8bf*)Vp)[i];
#endif

  // ---- phase 1: S = q @ k^T (pre-scaled q), 16 x 1024 into LDS ----
  v16bf qa0 = load_a_bf16(Qp, HD_, lane);        // d 0..31
  v16bf qa1 = load_a_bf16(Qp + 32, HD_, lane);   // d 32..63
  for (int kt = wave; kt < N_ / 16; kt += 8) {
    v8f acc = {};
    v16bf b0 = load_bT(Kp + (size_t)kt * 16 * HD_, HD_, lane);       // d 0..31
    v16bf b1 = load_bT(Kp + (size_t)kt * 16 * HD_ + 32, HD_, lane);  // d 32..63
    acc = wmma_bf16(qa0, b0, acc);
    acc = wmma_bf16(qa1, b1, acc);
#pragma unroll
    for (int r = 0; r < 8; ++r)
      sS[(r + ((lane >> 4) << 3)) * N_ + kt * 16 + (lane & 15)] = acc[r];
  }
  __syncthreads();

  // ---- phase 2: per-row top-k radix select + nucleus threshold + softmax ----
  for (int rr = 0; rr < 2; ++rr) {               // each wave owns 2 rows
    const int ri = wave * 2 + rr;
    const float* row = sS + (size_t)ri * N_;
    __bf16* prow = sP + (size_t)ri * N_;
    float vals[32];
    unsigned keys[32];
#pragma unroll
    for (int t = 0; t < 32; ++t) {
      float v = row[lane + t * 32];
      vals[t] = v;
      keys[t] = f2ord(v);
    }
    float m = vals[0];
#pragma unroll
    for (int t = 1; t < 32; ++t) m = fmaxf(m, vals[t]);
    for (int o = 16; o; o >>= 1) m = fmaxf(m, __shfl_xor(m, o));

    // radix-select the TOPK-th largest key (MSB-first)
    unsigned prefix = 0u;
    int target = TOPK_;
    for (int bit = 31; bit >= 0; --bit) {
      unsigned cand = prefix | (1u << bit);
      int c = 0;
#pragma unroll
      for (int t = 0; t < 32; ++t) c += ((keys[t] >> bit) == (cand >> bit));
      for (int o = 16; o; o >>= 1) c += __shfl_xor(c, o);
      if (c >= target) prefix = cand; else target -= c;
    }
    const unsigned kthKey = prefix;

    // softmax mass over kept set + compact kept values into LDS
    float Z = 0.f;
    int myCnt = 0;
#pragma unroll
    for (int t = 0; t < 32; ++t)
      if (keys[t] >= kthKey) { Z += __expf(vals[t] - m); ++myCnt; }
    for (int o = 16; o; o >>= 1) Z += __shfl_xor(Z, o);
    int off = myCnt;
    for (int o = 1; o < 32; o <<= 1) { int tu = __shfl_up(off, o); if (lane >= o) off += tu; }
    int total = __shfl(off, 31);
    off -= myCnt;
    if (total > 128) total = 128;
    int w = off;
#pragma unroll
    for (int t = 0; t < 32; ++t)
      if (keys[t] >= kthKey) { if (w < 128) sKept[wave][w] = vals[t]; ++w; }
    __syncthreads();

    // nucleus: thr = min kept value whose exclusive softmax mass <= TOP_P
    float thr = 3.0e38f;
    const float pz = TOPP_ * Z;
    for (int c = lane; c < total; c += 32) {
      float tv = sKept[wave][c];
      float em = 0.f;
      for (int s = 0; s < total; ++s) {
        float sv = sKept[wave][s];
        em += (sv > tv) ? __expf(sv - m) : 0.f;
      }
      if (em <= pz) thr = fminf(thr, tv);
    }
    for (int o = 16; o; o >>= 1) thr = fminf(thr, __shfl_xor(thr, o));

    // final masked softmax with fused NKAT row modulation, written as bf16
    float denom = 0.f;
#pragma unroll
    for (int t = 0; t < 32; ++t)
      if (vals[t] >= thr) denom += __expf(vals[t] - m);
    for (int o = 16; o; o >>= 1) denom += __shfl_xor(denom, o);
    const float sc = rowScale[b * N_ + q0 + ri] / denom;
#pragma unroll
    for (int t = 0; t < 32; ++t)
      prow[lane + t * 32] = f2bf((vals[t] >= thr) ? __expf(vals[t] - m) * sc : 0.f);
    __syncthreads();
  }

#if HAVE_TDM
  if (wave == 0) __builtin_amdgcn_s_wait_tensorcnt(0);   // V^T panel DMA complete
#endif
  __syncthreads();

  // ---- phase 3: out(16x64) = P(16x1024) @ V(1024x64); P and V^T in LDS ----
  if (wave < 4) {
    v8f acc = {};
    for (int k0 = 0; k0 < N_; k0 += 32) {
      v16bf a = load_a_bf16(sP + k0, N_, lane);                       // probs, bf16
      v16bf bf = load_bT(sVT + (size_t)(wave * 16) * N_ + k0, N_, lane);
      acc = wmma_bf16(a, bf, acc);
    }
#pragma unroll
    for (int r = 0; r < 8; ++r) {
      int mrow = r + ((lane >> 4) << 3);
      int d = wave * 16 + (lane & 15);
      AOUT[((size_t)(b * N_ + q0 + mrow) * C_) + h * HD_ + d] = f2bf(acc[r]);
    }
  }
}

// ---------------- output projection GEMM (2048x1024, K=1024) + bias ----------------
__global__ void __launch_bounds__(256) outproj_gemm_kernel(const __bf16* __restrict__ A,
                                                           const __bf16* __restrict__ W,
                                                           const float* __restrict__ bias,
                                                           float* __restrict__ Out) {
  __shared__ __align__(16) __bf16 sBT[128 * 32];
  const int lane = threadIdx.x & 31;
  const int wave = threadIdx.x >> 5;
  const int m0 = blockIdx.y * 128 + (wave >> 1) * 32;
  const int nblk = blockIdx.x * 128;
  const int nloc = (wave & 1) * 64;
  v8f acc[2][4] = {};
  for (int k0 = 0; k0 < C_; k0 += 32) {
    stage_bT(W, C_, k0, nblk, sBT, threadIdx.x);
    if (k0 + 32 < C_)
      __builtin_prefetch(W + (size_t)(k0 + 32 + ((threadIdx.x >> 4) << 1)) * C_
                         + nblk + ((threadIdx.x & 15) << 3), 0, 1);
    __syncthreads();
    v16bf a0 = load_a_bf16(A + (size_t)m0 * C_ + k0, C_, lane);
    v16bf a1 = load_a_bf16(A + (size_t)(m0 + 16) * C_ + k0, C_, lane);
#pragma unroll
    for (int j = 0; j < 4; ++j) {
      v16bf bj = load_bT(sBT + (nloc + j * 16) * 32, 32, lane);
      acc[0][j] = wmma_bf16(a0, bj, acc[0][j]);
      acc[1][j] = wmma_bf16(a1, bj, acc[1][j]);
    }
    __syncthreads();
  }
#pragma unroll
  for (int i = 0; i < 2; ++i)
#pragma unroll
    for (int j = 0; j < 4; ++j)
#pragma unroll
      for (int r = 0; r < 8; ++r) {
        int gm = m0 + i * 16 + r + ((lane >> 4) << 3);
        int gc = nblk + nloc + j * 16 + (lane & 15);
        Out[(size_t)gm * C_ + gc] = acc[i][j][r] + bias[gc];
      }
}

extern "C" void kernel_launch(void* const* d_in, const int* in_sizes, int n_in,
                              void* d_out, int out_size, void* d_ws, size_t ws_size,
                              hipStream_t stream) {
  const float* x      = (const float*)d_in[0];
  const float* w_qkv  = (const float*)d_in[1];
  const float* w_out  = (const float*)d_in[2];
  const float* b_out  = (const float*)d_in[3];
  const float* alpha  = (const float*)d_in[4];
  const float* beta   = (const float*)d_in[5];
  float* out = (float*)d_out;

  char* ws = (char*)d_ws;
  size_t off = 0;
  auto alloc = [&](size_t bytes) -> void* {
    void* p = ws + off;
    off += (bytes + 255) & ~(size_t)255;
    return p;
  };
  __bf16* xb     = (__bf16*)alloc((size_t)B_ * N_ * C_ * 2);
  __bf16* wqkvb  = (__bf16*)alloc((size_t)C_ * 3 * C_ * 2);
  __bf16* woutb  = (__bf16*)alloc((size_t)C_ * C_ * 2);
  __bf16* Qh     = (__bf16*)alloc((size_t)B_ * H_ * N_ * HD_ * 2);
  __bf16* Kh     = (__bf16*)alloc((size_t)B_ * H_ * N_ * HD_ * 2);
  __bf16* Vth    = (__bf16*)alloc((size_t)B_ * H_ * N_ * HD_ * 2);   // V^T per head
  __bf16* aout   = (__bf16*)alloc((size_t)B_ * N_ * C_ * 2);
  float*  rscale = (float*)alloc((size_t)B_ * N_ * 4);

  f32_to_bf16_kernel<<<2048, 256, 0, stream>>>(x, xb, B_ * N_ * C_);
  f32_to_bf16_kernel<<<2048, 256, 0, stream>>>(w_qkv, wqkvb, C_ * 3 * C_);
  f32_to_bf16_kernel<<<2048, 256, 0, stream>>>(w_out, woutb, C_ * C_);
  theta_kernel<<<B_ * N_, 256, 0, stream>>>(x, alpha, beta, rscale);
  qkv_gemm_kernel<<<dim3(3 * C_ / 128, (B_ * N_) / 128), 256, 0, stream>>>(xb, wqkvb, Qh, Kh, Vth);
  const size_t attn_lds = (size_t)16 * N_ * 4    // f32 scores
                        + (size_t)16 * N_ * 2    // bf16 probs
                        + (size_t)N_ * HD_ * 2;  // bf16 V^T panel
  attn_kernel<<<dim3(N_ / 16, B_ * H_), 256, attn_lds, stream>>>(Qh, Kh, Vth, rscale, aout);
  outproj_gemm_kernel<<<dim3(C_ / 128, (B_ * N_) / 128), 256, 0, stream>>>(aout, woutb, b_out, out);
}
